// _NETWORK_70858370450107
// MI455X (gfx1250) — matrix operand order repaired and checked by reading, not compiled, not run on hardware
//
#include <hip/hip_runtime.h>

// ---------------------------------------------------------------------------
// Types for CDNA5 WMMA (gfx1250, wave32)
// ---------------------------------------------------------------------------
typedef __attribute__((ext_vector_type(16))) __bf16 v16bf;
typedef __attribute__((ext_vector_type(8)))  __bf16 v8bf;
typedef __attribute__((ext_vector_type(8)))  float  v8f;

#define Bn 16
#define Sn 128
#define Vn 32000
#define Hn 512
#define En 512
#define Ln 256
#define Cn 512
#define G3 1536          // 3*H
#define BS 2048          // B*S

__device__ __forceinline__ __bf16 f2bf(float f) {
    unsigned u = __builtin_bit_cast(unsigned, f);
    u += 0x7FFFu + ((u >> 16) & 1u);          // round-to-nearest-even
    unsigned short s = (unsigned short)(u >> 16);
    return __builtin_bit_cast(__bf16, s);
}

// ---------------------------------------------------------------------------
// f32 -> bf16 conversion
// ---------------------------------------------------------------------------
__global__ __launch_bounds__(256) void k_f32_to_bf16(const float* __restrict__ in,
                                                     __bf16* __restrict__ out, int n) {
    int i = blockIdx.x * 256 + threadIdx.x;
    if (i < n) out[i] = f2bf(in[i]);
}

// ---------------------------------------------------------------------------
// WMMA GEMM:  C[M,N] = A[M,K](bf16) @ W[N,K](bf16)^T + bias[N]
// K == 512 (multiple of 32), M,N multiples of 128 per grid config (M=128 ok).
// Block: 128 threads = 4 waves in 2x2 -> 128x128 block tile; 64x64 per wave.
// ---------------------------------------------------------------------------
__global__ __launch_bounds__(128) void k_gemm_bf16(
    const __bf16* __restrict__ A, const __bf16* __restrict__ W,
    const float* __restrict__ bias, float* __restrict__ C,
    int M, int N, int K)
{
    const int lane   = threadIdx.x & 31;
    const int wave   = threadIdx.x >> 5;        // 0..3
    const int lane16 = lane & 15;
    const int hi     = lane >> 4;               // 0 or 1

    const int m0 = blockIdx.y * 128 + (wave >> 1) * 64;
    const int n0 = blockIdx.x * 128 + (wave & 1) * 64;

    const v8f zacc = {};
    v8f acc[4][4];
#pragma unroll
    for (int i = 0; i < 4; ++i)
#pragma unroll
        for (int j = 0; j < 4; ++j) acc[i][j] = zacc;

    for (int k0 = 0; k0 < K; k0 += 32) {
        v16bf a[4], b[4];
#pragma unroll
        for (int i = 0; i < 4; ++i) {
            const __bf16* arow = A + (size_t)(m0 + i * 16 + lane16) * K + k0 + 8 * hi;
            union { v16bf v; v8bf h[2]; } u;
            u.h[0] = *(const v8bf*)(arow);
            u.h[1] = *(const v8bf*)(arow + 16);
            a[i] = u.v;
        }
#pragma unroll
        for (int j = 0; j < 4; ++j) {
            const __bf16* brow = W + (size_t)(n0 + j * 16 + lane16) * K + k0 + 8 * hi;
            union { v16bf v; v8bf h[2]; } ub;
            ub.h[0] = *(const v8bf*)(brow);
            ub.h[1] = *(const v8bf*)(brow + 16);
            b[j] = ub.v;
        }
#pragma unroll
        for (int i = 0; i < 4; ++i)
#pragma unroll
            for (int j = 0; j < 4; ++j)
                acc[i][j] = __builtin_amdgcn_wmma_f32_16x16x32_bf16(
                    false, a[i], false, b[j], (short)0, acc[i][j], false, false);
    }

#pragma unroll
    for (int i = 0; i < 4; ++i) {
#pragma unroll
        for (int j = 0; j < 4; ++j) {
            const int ncol = n0 + j * 16 + lane16;
            const float bz = bias[ncol];
#pragma unroll
            for (int r = 0; r < 8; ++r) {
                const int mrow = m0 + i * 16 + r + 8 * hi;
                C[(size_t)mrow * N + ncol] = acc[i][j][r] + bz;
            }
        }
    }
}

// ---------------------------------------------------------------------------
// Embedding gather -> bf16 x (forward and time-reversed copies)
// ---------------------------------------------------------------------------
__global__ __launch_bounds__(256) void k_embed(const int* __restrict__ reviews,
                                               const float* __restrict__ emb,
                                               __bf16* __restrict__ xbf,
                                               __bf16* __restrict__ xrevbf) {
    int idx = blockIdx.x * 256 + threadIdx.x;             // B*S*E
    if (idx >= BS * En) return;
    int e  = idx & (En - 1);
    int bs = idx >> 9;
    int b  = bs >> 7, s = bs & (Sn - 1);
    int tok = reviews[bs];
    __bf16 v = f2bf(emb[(size_t)tok * En + e]);
    xbf[idx] = v;
    xrevbf[((size_t)(((Bn - 1 - b) << 7) | s) << 9) | e] = v;
}

// Decoder input: x = emb[reviews] + de_h[b] broadcast over s, as bf16
__global__ __launch_bounds__(256) void k_dec_input(const int* __restrict__ reviews,
                                                   const float* __restrict__ emb,
                                                   const float* __restrict__ de_h,
                                                   __bf16* __restrict__ xbf) {
    int idx = blockIdx.x * 256 + threadIdx.x;
    if (idx >= BS * En) return;
    int e  = idx & (En - 1);
    int bs = idx >> 9;
    int b  = bs >> 7;
    int tok = reviews[bs];
    xbf[idx] = f2bf(emb[(size_t)tok * En + e] + de_h[b * En + e]);
}

// ---------------------------------------------------------------------------
// GRU gate update (one scan step):
//   r = sig(ir+hr); z = sig(iz+hz); n = tanh(in + r*hn); h = (1-z)*n + z*h
// gi/gh rows: (128, 1536).  Writes h (f32 + bf16), optional ys (f32 / bf16).
// ---------------------------------------------------------------------------
__global__ __launch_bounds__(256) void k_gru_gate(const float* __restrict__ gi,
                                                  const float* __restrict__ gh,
                                                  float* __restrict__ h_f32,
                                                  __bf16* __restrict__ h_bf16,
                                                  float* __restrict__ ys_f32,
                                                  __bf16* __restrict__ ys_bf16) {
    int idx = blockIdx.x * 256 + threadIdx.x;             // 128*512
    if (idx >= Sn * Hn) return;
    int n = idx >> 9, h = idx & (Hn - 1);
    const float ir = gi[n * G3 + h];
    const float iz = gi[n * G3 + Hn + h];
    const float in_ = gi[n * G3 + 2 * Hn + h];
    const float hr = gh[n * G3 + h];
    const float hz = gh[n * G3 + Hn + h];
    const float hn = gh[n * G3 + 2 * Hn + h];
    const float r = 1.0f / (1.0f + expf(-(ir + hr)));
    const float z = 1.0f / (1.0f + expf(-(iz + hz)));
    const float nn = tanhf(in_ + r * hn);
    const float hv = (1.0f - z) * nn + z * h_f32[idx];
    h_f32[idx] = hv;
    h_bf16[idx] = f2bf(hv);
    if (ys_f32)  ys_f32[idx]  = hv;
    if (ys_bf16) ys_bf16[idx] = f2bf(hv);
}

// ---------------------------------------------------------------------------
// Encoder tail: gather last = [ys_f[b, len-1], ys_b[15-b, len-1]] then h2l
// hid[b,l] = last . h2l_W[l,:] + h2l_b[l]
// ---------------------------------------------------------------------------
__global__ __launch_bounds__(256) void k_enc_h2l(const float* __restrict__ ysf,
                                                 const float* __restrict__ ysb,
                                                 const int* __restrict__ lens,
                                                 const float* __restrict__ Wl,
                                                 const float* __restrict__ bl,
                                                 float* __restrict__ hid) {
    int idx = blockIdx.x * 256 + threadIdx.x;             // 16*256
    if (idx >= Bn * Ln) return;
    int b = idx >> 8, l = idx & (Ln - 1);
    int sf = lens[b] - 1;
    const float* rf = ysf + (size_t)(b * Sn + sf) * Hn;
    const float* rb = ysb + (size_t)((Bn - 1 - b) * Sn + sf) * Hn;
    const float* w  = Wl + (size_t)l * (2 * Hn);
    float acc = bl[l];
    for (int k = 0; k < Hn; ++k) acc += rf[k] * w[k];
    for (int k = 0; k < Hn; ++k) acc += rb[k] * w[Hn + k];
    hid[idx] = acc;
}

// ---------------------------------------------------------------------------
// VQ: argmin_c ||hid_b - E[:,c]||^2 (tie -> lowest c, matching argmax(-dist)),
// then quant[b,:] = E[:,c*]
// ---------------------------------------------------------------------------
__global__ __launch_bounds__(256) void k_vq(const float* __restrict__ hid,
                                            const float* __restrict__ Ecb,
                                            float* __restrict__ quant) {
    __shared__ float sh[Ln];
    __shared__ float sd[256];
    __shared__ int   si[256];
    const int b = blockIdx.x, t = threadIdx.x;
    sh[t] = hid[b * Ln + t];
    __syncthreads();
    float best = 3.4e38f; int besti = 1 << 30;
    for (int c = t; c < Cn; c += 256) {
        float d = 0.0f;
        for (int l = 0; l < Ln; ++l) {
            float diff = sh[l] - Ecb[(size_t)l * Cn + c];
            d += diff * diff;
        }
        if (d < best || (d == best && c < besti)) { best = d; besti = c; }
    }
    sd[t] = best; si[t] = besti;
    __syncthreads();
    for (int s = 128; s > 0; s >>= 1) {
        if (t < s) {
            if (sd[t + s] < sd[t] || (sd[t + s] == sd[t] && si[t + s] < si[t])) {
                sd[t] = sd[t + s]; si[t] = si[t + s];
            }
        }
        __syncthreads();
    }
    const int cbest = si[0];
    quant[b * Ln + t] = Ecb[(size_t)t * Cn + cbest];
}

// commit = 0.25 * mean((hid - quant)^2)
__global__ __launch_bounds__(256) void k_commit(const float* __restrict__ hid,
                                                const float* __restrict__ quant,
                                                float* __restrict__ outp) {
    __shared__ float s[256];
    const int t = threadIdx.x;
    float a = 0.0f;
    for (int i = t; i < Bn * Ln; i += 256) { float d = hid[i] - quant[i]; a += d * d; }
    s[t] = a;
    __syncthreads();
    for (int k = 128; k > 0; k >>= 1) { if (t < k) s[t] += s[t + k]; __syncthreads(); }
    if (t == 0) outp[0] = 0.25f * s[0] / (float)(Bn * Ln);
}

// de_h[b,e] = tanh(latent[b,:]) . l2o_W[e,:] + l2o_b[e]
__global__ __launch_bounds__(256) void k_l2o(const float* __restrict__ latent,
                                             const float* __restrict__ W,
                                             const float* __restrict__ bv,
                                             float* __restrict__ de_h) {
    int idx = blockIdx.x * 256 + threadIdx.x;             // 16*512
    if (idx >= Bn * En) return;
    int b = idx >> 9, e = idx & (En - 1);
    const float* w   = W + (size_t)e * Ln;
    const float* lat = latent + b * Ln;
    float acc = bv[e];
    for (int l = 0; l < Ln; ++l) acc += tanhf(lat[l]) * w[l];
    de_h[idx] = acc;
}

// ---------------------------------------------------------------------------
// Host orchestration
// ---------------------------------------------------------------------------
extern "C" void kernel_launch(void* const* d_in, const int* in_sizes, int n_in,
                              void* d_out, int out_size, void* d_ws, size_t ws_size,
                              hipStream_t stream) {
    (void)in_sizes; (void)n_in; (void)out_size; (void)ws_size;

    // ---- inputs (setup_inputs dict order, pytree-flattened) ----
    const int*   reviews  = (const int*)  d_in[0];
    const int*   lens     = (const int*)  d_in[1];
    const float* emb      = (const float*)d_in[4];
    const float* codebook = (const float*)d_in[5];        // (L=256, C=512)
    // enc_user: fwd(Wi,Wh,bi,bh)=6..9  bwd=10..13  h2l_W=14 h2l_b=15
    // enc_item: fwd=16..19 bwd=20..23 h2l_W=24 h2l_b=25
    // dec_user: l2o_W=26 l2o_b=27 gru=28..31
    // dec_item: l2o_W=32 l2o_b=33 gru=34..37
    const float* out_W = (const float*)d_in[38];
    const float* out_b = (const float*)d_in[39];

    // ---- workspace carve-up ----
    char* base = (char*)d_ws;
    size_t off = 0;
    auto alloc = [&](size_t bytes) -> char* {
        char* p = base + off;
        off = (off + bytes + 255) & ~(size_t)255;
        return p;
    };
    const size_t WIH = (size_t)G3 * Hn;                   // 786432
    __bf16* wbf[12];
    for (int i = 0; i < 12; ++i) wbf[i] = (__bf16*)alloc(WIH * 2);
    __bf16* outw_bf = (__bf16*)alloc((size_t)Vn * Hn * 2);
    __bf16* x_bf    = (__bf16*)alloc((size_t)BS * En * 2);
    __bf16* xrev_bf = (__bf16*)alloc((size_t)BS * En * 2);
    __bf16* xd_bf   = (__bf16*)alloc((size_t)BS * En * 2);
    float*  gi_all  = (float*) alloc((size_t)BS * G3 * 4);
    float*  gh      = (float*) alloc((size_t)Sn * G3 * 4);
    float*  h_f32   = (float*) alloc((size_t)Sn * Hn * 4);
    __bf16* h_bf    = (__bf16*)alloc((size_t)Sn * Hn * 2);
    float*  ysA     = (float*) alloc((size_t)BS * Hn * 4);
    float*  ysB     = (float*) alloc((size_t)BS * Hn * 4);
    __bf16* ys_du   = (__bf16*)alloc((size_t)BS * Hn * 2);
    __bf16* ys_di   = (__bf16*)alloc((size_t)BS * Hn * 2);
    float*  hid_u   = (float*) alloc((size_t)Bn * Ln * 4);
    float*  hid_i   = (float*) alloc((size_t)Bn * Ln * 4);
    float*  quant   = (float*) alloc((size_t)Bn * Ln * 4);
    float*  de_h    = (float*) alloc((size_t)Bn * En * 4);

    float* user_logits = (float*)d_out;
    float* item_logits = user_logits + (size_t)BS * Vn;
    float* commit_out  = item_logits + (size_t)BS * Vn;

    // ---- convert all GEMM weights to bf16 (once per call) ----
    const float* wsrc[12] = {
        (const float*)d_in[6],  (const float*)d_in[7],    // enc_user fwd
        (const float*)d_in[10], (const float*)d_in[11],   // enc_user bwd
        (const float*)d_in[16], (const float*)d_in[17],   // enc_item fwd
        (const float*)d_in[20], (const float*)d_in[21],   // enc_item bwd
        (const float*)d_in[28], (const float*)d_in[29],   // dec_user gru
        (const float*)d_in[34], (const float*)d_in[35]    // dec_item gru
    };
    for (int i = 0; i < 12; ++i)
        k_f32_to_bf16<<<(int)((WIH + 255) / 256), 256, 0, stream>>>(wsrc[i], wbf[i], (int)WIH);
    k_f32_to_bf16<<<(int)(((size_t)Vn * Hn + 255) / 256), 256, 0, stream>>>(
        out_W, outw_bf, Vn * Hn);

    // ---- embedding lookup (fwd + time-reversed) ----
    k_embed<<<(BS * En + 255) / 256, 256, 0, stream>>>(reviews, emb, x_bf, xrev_bf);

    // ---- GRU driver ----
    auto run_gru = [&](const __bf16* xbf, const __bf16* wi, const __bf16* wh,
                       const float* bi, const float* bh,
                       float* ysf, __bf16* ysbf) {
        // gi for all 16 scan steps at once: (2048,1536) = x @ Wi^T + bi
        dim3 g1(G3 / 128, BS / 128);
        k_gemm_bf16<<<g1, 128, 0, stream>>>(xbf, wi, bi, gi_all, BS, G3, Hn);
        hipMemsetAsync(h_f32, 0, (size_t)Sn * Hn * 4, stream);
        hipMemsetAsync(h_bf,  0, (size_t)Sn * Hn * 2, stream);
        dim3 g2(G3 / 128, 1);                     // M=128 fits one block row
        for (int t = 0; t < Bn; ++t) {
            k_gemm_bf16<<<g2, 128, 0, stream>>>(h_bf, wh, bh, gh, Sn, G3, Hn);
            k_gru_gate<<<(Sn * Hn + 255) / 256, 256, 0, stream>>>(
                gi_all + (size_t)t * Sn * G3, gh, h_f32, h_bf,
                ysf  ? ysf  + (size_t)t * Sn * Hn : nullptr,
                ysbf ? ysbf + (size_t)t * Sn * Hn : nullptr);
        }
    };

    // ---- user encoder ----
    run_gru(x_bf,    wbf[0], wbf[1], (const float*)d_in[8],  (const float*)d_in[9],  ysA, nullptr);
    run_gru(xrev_bf, wbf[2], wbf[3], (const float*)d_in[12], (const float*)d_in[13], ysB, nullptr);
    k_enc_h2l<<<(Bn * Ln + 255) / 256, 256, 0, stream>>>(
        ysA, ysB, lens, (const float*)d_in[14], (const float*)d_in[15], hid_u);

    // ---- VQ codebook lookup + commit loss ----
    k_vq<<<Bn, 256, 0, stream>>>(hid_u, codebook, quant);
    k_commit<<<1, 256, 0, stream>>>(hid_u, quant, commit_out);

    // ---- item encoder ----
    run_gru(x_bf,    wbf[4], wbf[5], (const float*)d_in[18], (const float*)d_in[19], ysA, nullptr);
    run_gru(xrev_bf, wbf[6], wbf[7], (const float*)d_in[22], (const float*)d_in[23], ysB, nullptr);
    k_enc_h2l<<<(Bn * Ln + 255) / 256, 256, 0, stream>>>(
        ysA, ysB, lens, (const float*)d_in[24], (const float*)d_in[25], hid_i);

    // ---- user decoder (latent = quantized user code) ----
    k_l2o<<<(Bn * En + 255) / 256, 256, 0, stream>>>(
        quant, (const float*)d_in[26], (const float*)d_in[27], de_h);
    k_dec_input<<<(BS * En + 255) / 256, 256, 0, stream>>>(reviews, emb, de_h, xd_bf);
    run_gru(xd_bf, wbf[8], wbf[9], (const float*)d_in[30], (const float*)d_in[31], nullptr, ys_du);

    // ---- item decoder (latent = item hidden) ----
    k_l2o<<<(Bn * En + 255) / 256, 256, 0, stream>>>(
        hid_i, (const float*)d_in[32], (const float*)d_in[33], de_h);
    k_dec_input<<<(BS * En + 255) / 256, 256, 0, stream>>>(reviews, emb, de_h, xd_bf);
    run_gru(xd_bf, wbf[10], wbf[11], (const float*)d_in[36], (const float*)d_in[37], nullptr, ys_di);

    // ---- vocab projections: (2048,512) @ (32000,512)^T + out_b ----
    dim3 gv(Vn / 128, BS / 128);                  // 250 x 16
    k_gemm_bf16<<<gv, 128, 0, stream>>>(ys_du, outw_bf, out_b, user_logits, BS, Vn, Hn);
    k_gemm_bf16<<<gv, 128, 0, stream>>>(ys_di, outw_bf, out_b, item_logits, BS, Vn, Hn);
}
